// SelfAttention_43465069035935
// MI455X (gfx1250) — matrix operand order repaired and checked by reading, not compiled, not run on hardware
//
#include <hip/hip_runtime.h>

// ---------------------------------------------------------------------------
// Flash-attention for B=8, S=2048, D=512 (fp32 in/out) on gfx1250 (MI455X).
//  - f16 WMMA (v_wmma_f32_16x16x32_f16), fp32 accumulation, online softmax.
//  - Pre-pass converts K -> f16 [b][s][d] and V -> f16 transposed [b][d][s].
//  - K tiles staged by the Tensor Data Mover (tensor_load_to_lds, TENSORcnt),
//    using TDM's pad feature to produce the bank-conflict-free LDS stride.
//  - V tiles staged with GLOBAL_LOAD_ASYNC_TO_LDS_B128 (ASYNCcnt).
//  - Double-buffered K: K(j+1) DMA overlaps softmax+PV(j); V(j) overlaps QK(j).
//  - CTA: 256 threads (8 waves), 64 query rows. Wave (r = w&3, c = w>>2):
//      QK^T split over D halves (c), partials reduced via LDS;
//      softmax by c==0 waves; PV split over output-D halves (c).
// ---------------------------------------------------------------------------

typedef _Float16     v16h  __attribute__((ext_vector_type(16)));
typedef float        v8f   __attribute__((ext_vector_type(8)));
typedef unsigned int u32x4 __attribute__((ext_vector_type(4)));
typedef int          i32x8 __attribute__((ext_vector_type(8)));
typedef int          i32x4 __attribute__((ext_vector_type(4)));
typedef float        f32x4 __attribute__((ext_vector_type(4)));
typedef _Float16     h4    __attribute__((ext_vector_type(4)));

#define Bb 8
#define Ss 2048
#define Dd 512
#define BM 64          // query rows per CTA
#define BN 64          // keys per KV tile
#define KPAD 528       // sK row stride in halves (= 512 + 16; TDM pad 8 dwords)
#define VPAD 80        // sVt row stride in halves (= 64 + 16)
#define PPAD 80        // sP  row stride in halves
#define NT  (Ss / BN)  // 32 KV tiles

#if __has_builtin(__builtin_amdgcn_tensor_load_to_lds) && \
    __has_builtin(__builtin_amdgcn_s_wait_tensorcnt)
#define USE_TDM_K 1
#else
#define USE_TDM_K 0
#endif

union V16 { v16h h; u32x4 q[2]; };

static __device__ inline v8f vzero8() {
  v8f z;
#pragma unroll
  for (int i = 0; i < 8; ++i) z[i] = 0.0f;
  return z;
}

static __device__ inline void async_copy_b128(const _Float16* src, _Float16* ldsDst) {
  // GV mode: 64-bit global address in a VGPR pair, LDS byte offset in a VGPR
  // (low 32 bits of the generic pointer = LDS offset per FLAT->LDS truncation).
  unsigned lds = (unsigned)(unsigned long long)(void*)ldsDst;
  unsigned long long ga = (unsigned long long)(const void*)src;
  asm volatile("global_load_async_to_lds_b128 %0, %1, off"
               :: "v"(lds), "v"(ga) : "memory");
}

static __device__ inline void wait_async_zero() {
#if __has_builtin(__builtin_amdgcn_s_wait_asynccnt)
  __builtin_amdgcn_s_wait_asynccnt(0);
#else
  asm volatile("s_wait_asynccnt 0" ::: "memory");
#endif
}

#if USE_TDM_K
// TDM: load a BN x Dd (f16) tile whose rows are Dd halves apart in memory into
// LDS with row stride KPAD halves (pad 8 dwords after every 256 dwords).
static __device__ inline void tdm_stage_k(const _Float16* src, _Float16* ldsDst) {
  unsigned long long ga = (unsigned long long)(const void*)src;
  unsigned lds = (unsigned)(unsigned long long)(void*)ldsDst;

  u32x4 g0;
  g0[0] = 1u;                                           // count=1, user mode
  g0[1] = lds;                                          // lds_addr (bytes)
  g0[2] = (unsigned)(ga & 0xffffffffu);                 // global_addr[31:0]
  g0[3] = (unsigned)((ga >> 32) & 0x1ffffffu)           // global_addr[56:32]
        | (2u << 30);                                   // type = 2 ("image")

  i32x8 g1;
  g1[0] = (1 << 16)        // data_size = 1 -> 2 bytes
        | (1 << 20)        // pad_enable
        | (7 << 22)        // pad_interval: 256 dwords (one 1024B row)
        | (7 << 25);       // pad_amount: 8 dwords (32B) -> stride 1056B = KPAD*2
  g1[1] = (Dd & 0xffff) << 16;                 // abar=0 | tensor_dim0[15:0]
  g1[2] = (Dd >> 16) | ((Ss & 0xffff) << 16);  // tensor_dim0[31:16] | tensor_dim1[15:0]
  g1[3] = (Ss >> 16) | (Dd << 16);             // tensor_dim1[31:16] | tile_dim0 = Dd
  g1[4] = BN;                                  // tile_dim1 = BN | tile_dim2 = 0
  g1[5] = Dd;                                  // tensor_dim0_stride[31:0]
  g1[6] = 0;                                   // stride hi | tensor_dim1_stride lo
  g1[7] = 0;

  i32x4 gz4; gz4[0] = 0; gz4[1] = 0; gz4[2] = 0; gz4[3] = 0;
  i32x8 gz8;
#pragma unroll
  for (int i = 0; i < 8; ++i) gz8[i] = 0;
  // 6-arg toolchain form: (group0, group1, group2, group3, group4, cpol)
  __builtin_amdgcn_tensor_load_to_lds(g0, g1, gz4, gz4, gz8, 0);
}
#endif

// ---- pre-pass: K fp32 -> f16, same layout [b][s][d] --------------------
__global__ void cvt_k_kernel(const float* __restrict__ in, _Float16* __restrict__ out) {
  int i = (blockIdx.x * blockDim.x + threadIdx.x) * 4;
  f32x4 v = *(const f32x4*)(in + i);
  h4 o;
#pragma unroll
  for (int e = 0; e < 4; ++e) o[e] = (_Float16)v[e];
  *(h4*)(out + i) = o;
}

// ---- pre-pass: V fp32 [b][s][d] -> f16 transposed [b][d][s] ------------
__global__ void cvt_vt_kernel(const float* __restrict__ v, _Float16* __restrict__ out) {
  long long t = (long long)blockIdx.x * blockDim.x + threadIdx.x; // over [b][d][s]
  int s = (int)(t % Ss);
  long long r = t / Ss;
  int d = (int)(r % Dd);
  int b = (int)(r / Dd);
  out[t] = (_Float16)(v[((long long)(b * Ss + s)) * Dd + d]);
}

// ---- main flash-attention kernel ---------------------------------------
__launch_bounds__(256, 1)
__global__ void fa_kernel(const float* __restrict__ Q,
                          const _Float16* __restrict__ Kh,
                          const _Float16* __restrict__ Vt,
                          float* __restrict__ Out) {
  extern __shared__ char smem[];
  _Float16* sK0  = (_Float16*)smem;          // [BN][KPAD]   f16 key tile, buf 0
  _Float16* sK1  = sK0 + BN * KPAD;          // [BN][KPAD]   f16 key tile, buf 1
  _Float16* sVt  = sK1 + BN * KPAD;          // [Dd][VPAD]   f16 V^T tile
  _Float16* sP   = sVt + Dd * VPAD;          // [BM][PPAD]   f16 probabilities
  float*    sPart = (float*)(sP + BM * PPAD);// [BM][BN]     fp32 partial scores
  float*    sRow  = sPart + BM * BN;         // [BM]         alpha / final l

  const int tid  = threadIdx.x;
  const int lane = tid & 31;
  const int wid  = tid >> 5;
  const int rg   = wid & 3;   // row group (16 rows)
  const int ch   = wid >> 2;  // column half of D
  const int lm   = lane & 15;
  const int hi   = lane >> 4;
  const int b    = blockIdx.y;
  const int qbase = blockIdx.x * BM;

  const float scale = 0.044194173824159216f; // 1/sqrt(512)

  // ---- prologue: kick off K(0) stage ----
  {
    const _Float16* kt = Kh + ((long long)(b * Ss + 0 * BN)) * Dd;
#if USE_TDM_K
    if (wid == 0) tdm_stage_k(kt, sK0);
#else
#pragma unroll
    for (int i = 0; i < (BN * Dd / 8) / 256; ++i) {  // 16 iters
      int c = i * 256 + tid;
      int key = c >> 6;
      int off = (c & 63) * 8;
      async_copy_b128(kt + key * Dd + off, sK0 + key * KPAD + off);
    }
#endif
  }

  // ---- Q A-fragments: 16 rows x 256 cols (this wave's D half) ----------
  // 16-bit A 16x32 layout: lane m (0..15) row m, K = kk + hi*8 + e (e<8)
  //                        and K = kk + 16 + hi*8 + e for elements 8..15.
  v16h aQ[8];
  {
    const int qrow = qbase + rg * 16 + lm;
    const float* qp = Q + ((long long)(b * Ss + qrow)) * Dd;
#pragma unroll
    for (int f = 0; f < 8; ++f) {
      const int kk = ch * 256 + f * 32;
      f32x4 g0 = *(const f32x4*)(qp + kk + hi * 8);
      f32x4 g1 = *(const f32x4*)(qp + kk + hi * 8 + 4);
      f32x4 g2 = *(const f32x4*)(qp + kk + 16 + hi * 8);
      f32x4 g3 = *(const f32x4*)(qp + kk + 16 + hi * 8 + 4);
      v16h a;
#pragma unroll
      for (int e = 0; e < 4; ++e) {
        a[e]      = (_Float16)(g0[e] * scale);
        a[4 + e]  = (_Float16)(g1[e] * scale);
        a[8 + e]  = (_Float16)(g2[e] * scale);
        a[12 + e] = (_Float16)(g3[e] * scale);
      }
      aQ[f] = a;
    }
  }

  // ---- O accumulator: 16 rows x 256 cols fp32 (C-layout tiles) ---------
  v8f Oacc[16];
#pragma unroll
  for (int t = 0; t < 16; ++t) Oacc[t] = vzero8();

  float mrun[8], lrun[8];
#pragma unroll
  for (int r = 0; r < 8; ++r) { mrun[r] = -1e30f; lrun[r] = 0.0f; }

  // K(0) must be resident before the first QK
#if USE_TDM_K
  if (wid == 0) __builtin_amdgcn_s_wait_tensorcnt(0);
#else
  wait_async_zero();
#endif
  __syncthreads();

  for (int jt = 0; jt < NT; ++jt) {
    _Float16* sK = (jt & 1) ? sK1 : sK0;

    // ---- issue async V(jt) stage; overlaps with QK(jt) below ----
    {
      const _Float16* vt = Vt + (long long)b * Dd * Ss + jt * BN;
#pragma unroll
      for (int i = 0; i < (Dd * BN / 8) / 256; ++i) {  // 16 iters
        int c = i * 256 + tid;
        int d = c >> 3;
        int off = (c & 7) * 8;
        async_copy_b128(vt + (long long)d * Ss + off, sVt + d * VPAD + off);
      }
    }

    // ---- partial scores: Q(16x256half) @ K^T -> 16x64 fp32 ----
    // B 32x16 layout: lane n (0..15) col n, K = kk + hi*8 + e (e<8)
    // and K = kk + 16 + hi*8 + e for elements 8..15.
    v8f acc[4];
#pragma unroll
    for (int t = 0; t < 4; ++t) acc[t] = vzero8();
#pragma unroll
    for (int t = 0; t < 4; ++t) {
#pragma unroll
      for (int f = 0; f < 8; ++f) {
        const int kk = ch * 256 + f * 32;
        const _Float16* bp = sK + (t * 16 + lm) * KPAD + kk + hi * 8;
        V16 bf;
        bf.q[0] = *(const u32x4*)bp;
        bf.q[1] = *(const u32x4*)(bp + 16);
        acc[t] = __builtin_amdgcn_wmma_f32_16x16x32_f16(
            false, aQ[f], false, bf.h, (short)0, acc[t], false, false);
      }
    }

    // ---- reduce the two D-halves through LDS ----
    if (ch == 1) {
#pragma unroll
      for (int t = 0; t < 4; ++t)
#pragma unroll
        for (int r = 0; r < 8; ++r)
          sPart[(rg * 16 + r + hi * 8) * BN + t * 16 + lm] = acc[t][r];
    }
    wait_async_zero();   // V(jt) resident for everyone after this barrier
    __syncthreads();

    // ---- issue K(jt+1) prefetch; overlaps softmax + PV(jt) ----
    if (jt + 1 < NT) {
      _Float16* sKn = ((jt + 1) & 1) ? sK1 : sK0;
      const _Float16* kt = Kh + ((long long)(b * Ss + (jt + 1) * BN)) * Dd;
#if USE_TDM_K
      if (wid == 0) tdm_stage_k(kt, sKn);
#else
#pragma unroll
      for (int i = 0; i < (BN * Dd / 8) / 256; ++i) {  // 16 iters
        int c = i * 256 + tid;
        int key = c >> 6;
        int off = (c & 63) * 8;
        async_copy_b128(kt + key * Dd + off, sKn + key * KPAD + off);
      }
#endif
    }

    if (ch == 0) {
#pragma unroll
      for (int t = 0; t < 4; ++t)
#pragma unroll
        for (int r = 0; r < 8; ++r)
          acc[t][r] += sPart[(rg * 16 + r + hi * 8) * BN + t * 16 + lm];

      // ---- online softmax over this wave's 16 rows ----
      float alpha[8];
#pragma unroll
      for (int r = 0; r < 8; ++r) {
        float mx = fmaxf(fmaxf(acc[0][r], acc[1][r]), fmaxf(acc[2][r], acc[3][r]));
#pragma unroll
        for (int s = 1; s < 16; s <<= 1) mx = fmaxf(mx, __shfl_xor(mx, s, 32));
        float mnew = fmaxf(mrun[r], mx);
        alpha[r] = __expf(mrun[r] - mnew);
        mrun[r] = mnew;
        float ls = 0.0f;
#pragma unroll
        for (int t = 0; t < 4; ++t) {
          float p = __expf(acc[t][r] - mnew);
          acc[t][r] = p;
          ls += p;
        }
#pragma unroll
        for (int s = 1; s < 16; s <<= 1) ls += __shfl_xor(ls, s, 32);
        lrun[r] = lrun[r] * alpha[r] + ls;
      }
      // publish P (f16) and alpha
#pragma unroll
      for (int t = 0; t < 4; ++t)
#pragma unroll
        for (int r = 0; r < 8; ++r)
          sP[(rg * 16 + r + hi * 8) * PPAD + t * 16 + lm] = (_Float16)acc[t][r];
      if (lm == 0) {
#pragma unroll
        for (int r = 0; r < 8; ++r) sRow[rg * 16 + r + hi * 8] = alpha[r];
      }
    }
    __syncthreads();

    // ---- rescale O and accumulate P @ V over this wave's D half ----
    {
      float al[8];
#pragma unroll
      for (int r = 0; r < 8; ++r) al[r] = sRow[rg * 16 + r + hi * 8];
#pragma unroll
      for (int t = 0; t < 16; ++t)
#pragma unroll
        for (int r = 0; r < 8; ++r) Oacc[t][r] *= al[r];

      // P A-fragments: 16x64 -> two K=32 steps
      V16 pf[2];
#pragma unroll
      for (int s2 = 0; s2 < 2; ++s2) {
        const _Float16* pp = sP + (rg * 16 + lm) * PPAD + s2 * 32 + hi * 8;
        pf[s2].q[0] = *(const u32x4*)pp;
        pf[s2].q[1] = *(const u32x4*)(pp + 16);
      }
#pragma unroll
      for (int t = 0; t < 16; ++t) {
        const int dbase = ch * 256 + t * 16;
#pragma unroll
        for (int s2 = 0; s2 < 2; ++s2) {
          const _Float16* vp = sVt + (dbase + lm) * VPAD + s2 * 32 + hi * 8;
          V16 bv;
          bv.q[0] = *(const u32x4*)vp;
          bv.q[1] = *(const u32x4*)(vp + 16);
          Oacc[t] = __builtin_amdgcn_wmma_f32_16x16x32_f16(
              false, pf[s2].h, false, bv.h, (short)0, Oacc[t], false, false);
        }
      }
    }
    // K(jt+1) must be resident (and PV done) before next iteration
#if USE_TDM_K
    if (wid == 0) __builtin_amdgcn_s_wait_tensorcnt(0);
#else
    wait_async_zero();
#endif
    __syncthreads();
  }

  // ---- epilogue: normalize by l and store ----
  if (ch == 0 && lm == 0) {
#pragma unroll
    for (int r = 0; r < 8; ++r) sRow[rg * 16 + r + hi * 8] = lrun[r];
  }
  __syncthreads();
  {
    float linv[8];
#pragma unroll
    for (int r = 0; r < 8; ++r) linv[r] = 1.0f / sRow[rg * 16 + r + hi * 8];
#pragma unroll
    for (int t = 0; t < 16; ++t) {
#pragma unroll
      for (int r = 0; r < 8; ++r) {
        const int row = qbase + rg * 16 + r + hi * 8;
        const int col = ch * 256 + t * 16 + lm;
        Out[((long long)(b * Ss + row)) * Dd + col] = Oacc[t][r] * linv[r];
      }
    }
  }
}

extern "C" void kernel_launch(void* const* d_in, const int* in_sizes, int n_in,
                              void* d_out, int out_size, void* d_ws, size_t ws_size,
                              hipStream_t stream) {
  (void)in_sizes; (void)n_in; (void)out_size; (void)ws_size;
  const float* V = (const float*)d_in[0];  // "value"
  const float* K = (const float*)d_in[1];  // "key"
  const float* Q = (const float*)d_in[2];  // "query"
  float* Out = (float*)d_out;

  _Float16* wsK  = (_Float16*)d_ws;                       // [B][S][D] f16
  _Float16* wsVt = wsK + (size_t)Bb * Ss * Dd;            // [B][D][S] f16

  const long long N = (long long)Bb * Ss * Dd;            // 8.39M elements
  cvt_k_kernel<<<dim3((unsigned)(N / (256 * 4))), dim3(256), 0, stream>>>(K, wsK);
  cvt_vt_kernel<<<dim3((unsigned)(N / 256)), dim3(256), 0, stream>>>(V, wsVt);

  const size_t smem = (size_t)(2 * BN * KPAD + Dd * VPAD + BM * PPAD) * sizeof(_Float16)
                    + (size_t)(BM * BN) * sizeof(float)
                    + (size_t)BM * sizeof(float);
  fa_kernel<<<dim3(Ss / BM, Bb), dim3(256), smem, stream>>>(Q, wsK, wsVt, Out);
}
